// LambdaRankLoss_29703993819970
// MI455X (gfx1250) — 8-bit matrix operand order repaired and checked by reading, compiled, not run on hardware
//
#include <hip/hip_runtime.h>
#include <stdint.h>


typedef __attribute__((ext_vector_type(8))) int v8i;

#define BB 4
#define NN 64
#define TT 2048

__global__ void lam_zero(float* __restrict__ Zacc, float* __restrict__ out) {
    int idx = blockIdx.x * blockDim.x + threadIdx.x;
    if (idx < BB * NN * NN) Zacc[idx] = 0.0f;
    if (idx == 0) out[0] = 0.0f;
}

__global__ void lam_prep(const float* __restrict__ Msc,
                         const int* __restrict__ preds,
                         const float* __restrict__ emb,
                         uint8_t* __restrict__ tag8,
                         uint8_t* __restrict__ S8,
                         uint8_t* __restrict__ m8) {
    const int b = blockIdx.x;
    for (int t = threadIdx.x; t < TT; t += blockDim.x) {
        m8[b * TT + t] = (Msc[b * TT + t] != 0.0f) ? 1 : 0;
        unsigned run = 0;
        for (int kk = 0; kk < NN; ++kk) {
            int api = preds[b * NN + kk];
            unsigned v = (emb[(size_t)api * TT + t] != 0.0f) ? 1u : 0u;
            run |= v;
            tag8[((size_t)b * NN + kk) * TT + t] = (uint8_t)v;
            S8[((size_t)b * NN + kk) * TT + t] = (uint8_t)run;
        }
    }
}

__global__ __launch_bounds__(256) void lam_main(const uint8_t* __restrict__ tag8,
                                                const uint8_t* __restrict__ S8,
                                                const uint8_t* __restrict__ m8,
                                                float* __restrict__ Zacc,
                                                float* __restrict__ condArr) {
    const int k = blockIdx.x;
    const int b = blockIdx.y;
    const int tid = threadIdx.x;

    const uint8_t* tagB = tag8 + (size_t)b * NN * TT;
    const uint8_t* Srow = S8 + ((size_t)b * NN + k) * TT;
    const uint8_t* mrow = m8 + (size_t)b * TT;
    const unsigned O = 0x01010101u;

    __shared__ int sh_cnzM, sh_cntMS, sh_cntSgtM;
    if (tid == 0) { sh_cnzM = 0; sh_cntMS = 0; sh_cntSgtM = 0; }
    __syncthreads();
    {
        int t0b = tid * 8;
        unsigned m0 = *(const unsigned*)(mrow + t0b);
        unsigned m1 = *(const unsigned*)(mrow + t0b + 4);
        unsigned s0 = *(const unsigned*)(Srow + t0b);
        unsigned s1 = *(const unsigned*)(Srow + t0b + 4);
        int cm  = __popc(m0) + __popc(m1);
        int cms = __popc(m0 & (s0 ^ O)) + __popc(m1 & (s1 ^ O));
        int csm = __popc(s0 & (m0 ^ O)) + __popc(s1 & (m1 ^ O));
        atomicAdd(&sh_cnzM, cm);
        atomicAdd(&sh_cntMS, cms);
        atomicAdd(&sh_cntSgtM, csm);
    }
    __syncthreads();

    const int wave = tid >> 5;
    const int lane = tid & 31;
    const int half = lane >> 4;
    const int r = lane & 15;
    const int i0  = (wave >> 1) * 16;
    const int j0a = (wave & 1) * 32;
    const int j0b = j0a + 16;

    const uint8_t* rowAi = tagB + (size_t)(i0 + r) * TT;
    const uint8_t* rowJa = tagB + (size_t)(j0a + r) * TT;
    const uint8_t* rowJb = tagB + (size_t)(j0b + r) * TT;

    v8i c1a = {}, c1b = {};
    v8i qla = {}, qlb = {};
    v8i qza = {}, qzb = {};

    for (int t0 = 0; t0 < TT; t0 += 64) {
        v8i A1, A2, A3, B1a, B2a, B3a, B1b, B2b, B3b;
#pragma unroll
        for (int v = 0; v < 8; ++v) {
            int tA = t0 + ((v >> 1) * 16 + (v & 1) * 4) + half * 8;
            unsigned gi = *(const unsigned*)(rowAi + tA);
            unsigned sv = *(const unsigned*)(Srow + tA);
            unsigned mv = *(const unsigned*)(mrow + tA);
            unsigned nS = sv ^ O, nM = mv ^ O, nG = gi ^ O;
            A1[v] = (int)(mv & (nS | gi));
            A2[v] = (int)((sv & nG & nM) ^ O);
            A3[v] = (int)(nS | gi);
            int tB = t0 + ((v >> 1) * 16 + (v & 1) * 4) + half * 8;
            unsigned mb = *(const unsigned*)(mrow + tB);
            unsigned nMb = mb ^ O;
            unsigned ha = *(const unsigned*)(rowJa + tB);
            unsigned hb = *(const unsigned*)(rowJb + tB);
            B1a[v] = (int)(mb & (ha ^ O));
            B2a[v] = (int)((ha & nMb) ^ O);
            B3a[v] = (int)(ha ^ O);
            B1b[v] = (int)(mb & (hb ^ O));
            B2b[v] = (int)((hb & nMb) ^ O);
            B3b[v] = (int)(hb ^ O);
        }
        c1a = __builtin_amdgcn_wmma_i32_16x16x64_iu8(false, A1, false, B1a, c1a, false, false);
        c1b = __builtin_amdgcn_wmma_i32_16x16x64_iu8(false, A1, false, B1b, c1b, false, false);
        qla = __builtin_amdgcn_wmma_i32_16x16x64_iu8(false, A2, false, B2a, qla, false, false);
        qlb = __builtin_amdgcn_wmma_i32_16x16x64_iu8(false, A2, false, B2b, qlb, false, false);
        qza = __builtin_amdgcn_wmma_i32_16x16x64_iu8(false, A3, false, B3a, qza, false, false);
        qzb = __builtin_amdgcn_wmma_i32_16x16x64_iu8(false, A3, false, B3b, qzb, false, false);
    }

    const float cnzM    = (float)sh_cnzM;
    const float cntMS   = (float)sh_cntMS;
    const float cSgtM   = (float)sh_cntSgtM;
    const float invlogk = 1.0f / log2f((float)k + 2.0f);
    const float w1      = invlogk / (cnzM + 1.0f);
    const float t2base  = cSgtM * (1.0f / (float)TT);
    float* Zb = Zacc + (size_t)b * NN * NN;
    if (tid == 0) condArr[b * NN + k] = cSgtM;

#pragma unroll
    for (int v = 0; v < 8; ++v) {
        int i = i0 + half * 8 + v;
        if (i <= k) {
            int ja = j0a + r;
            if (ja > k) {
                float contrib = (cntMS - (float)c1a[v]) * w1 +
                    (t2base - ((float)TT - (float)qla[v]) /
                              ((float)TT + 1.0f - (float)qza[v])) * invlogk;
                atomicAdd(&Zb[i * NN + ja], contrib);
            }
            int jb = j0b + r;
            if (jb > k) {
                float contrib = (cntMS - (float)c1b[v]) * w1 +
                    (t2base - ((float)TT - (float)qlb[v]) /
                              ((float)TT + 1.0f - (float)qzb[v])) * invlogk;
                atomicAdd(&Zb[i * NN + jb], contrib);
            }
        }
    }
}

__global__ __launch_bounds__(256) void lam_final(const float* __restrict__ ys,
                                                 const float* __restrict__ Zacc,
                                                 const float* __restrict__ condArr,
                                                 float* __restrict__ out) {
    const int b = blockIdx.x;
    const int tid = threadIdx.x;
    __shared__ int s_first;
    __shared__ float s_sum;
    if (tid == 0) { s_first = NN; s_sum = 0.0f; }
    __syncthreads();
    if (tid < NN && condArr[b * NN + tid] == 0.0f) atomicMin(&s_first, tid);
    __syncthreads();
    const float mink = 1.0f / log2f(2.0f + (float)s_first);
    const float* Zb = Zacc + b * NN * NN;
    const float* y = ys + b * NN;
    float acc = 0.0f;
    for (int idx = tid; idx < NN * NN; idx += 256) {
        int i = idx >> 6, j = idx & 63;
        if (j > i) {
            float z = Zb[idx] / (float)(j - i + 1);
            float lam = 1.0f / (1.0f + expf(y[i] - y[j]));
            acc += lam * z;
        }
    }
    atomicAdd(&s_sum, acc);
    __syncthreads();
    if (tid == 0) {
        float s = s_sum * mink * (1.0f / (float)(NN + 1));
        atomicAdd(out, fmaxf(s, 0.0f) * (1.0f / (float)BB));
    }
}

extern "C" void kernel_launch(void* const* d_in, const int* in_sizes, int n_in,
                              void* d_out, int out_size, void* d_ws, size_t ws_size,
                              hipStream_t stream) {
    const float* y_scores = (const float*)d_in[0];
    const int*   y_preds  = (const int*)d_in[1];
    const float* Msc      = (const float*)d_in[2];
    const float* emb      = (const float*)d_in[3];
    float* out = (float*)d_out;

    uint8_t* ws = (uint8_t*)d_ws;
    uint8_t* tag8 = ws;
    uint8_t* S8   = ws + (size_t)BB * NN * TT;
    uint8_t* m8   = ws + 2 * (size_t)BB * NN * TT;
    float* condArr = (float*)(ws + 2 * (size_t)BB * NN * TT + BB * TT);
    float* Zacc    = (float*)(ws + 2 * (size_t)BB * NN * TT + BB * TT + 1024);

    lam_zero<<<dim3((BB * NN * NN + 255) / 256), dim3(256), 0, stream>>>(Zacc, out);
    lam_prep<<<dim3(BB), dim3(256), 0, stream>>>(Msc, y_preds, emb, tag8, S8, m8);
    lam_main<<<dim3(NN, BB), dim3(256), 0, stream>>>(tag8, S8, m8, Zacc, condArr);
    lam_final<<<dim3(BB), dim3(256), 0, stream>>>(y_scores, Zacc, condArr, out);
}
